// Net_78400333021808
// MI455X (gfx1250) — matrix operand order repaired and checked by reading, not compiled, not run on hardware
//
#include <hip/hip_runtime.h>
#include <hip/hip_bf16.h>

// CDNA5 / gfx1250 AGNN forward.
//  - lin1 GEMM uses v_wmma_f32_16x16x32_f16 (wave32, 16x16 tile per wave).
//  - edge propagation: half-wave (16 lanes) per edge; node tables (6.4MB) are
//    L2-resident on MI455X (192MB L2), so gathers/atomics stay on-chip.
//  - segment-max stabilizer omitted: alpha = beta*cos in [-|beta|,|beta|],
//    exp() is well-conditioned and the stabilizer cancels exactly.

typedef __attribute__((ext_vector_type(16))) _Float16 v16h;
typedef __attribute__((ext_vector_type(8)))  float    v8f;

// ---------------------------------------------------------------------------
// Kernel 1: h0 = relu(x @ W1^T + b1)   x:[n,512] W1:[16,512] -> h0:[n,16]
// One wave per 16-row tile. A = x tile (16x32 f16 per step), B = W1^T chunk.
// ---------------------------------------------------------------------------
__global__ void __launch_bounds__(256) k_lin1_wmma(
    const float* __restrict__ x,
    const float* __restrict__ w1,
    const float* __restrict__ b1,
    float* __restrict__ h0,
    int n_nodes)
{
    __shared__ _Float16 sw[16 * 512];   // W1 as f16, row-major [16][512]
    for (int i = threadIdx.x; i < 16 * 512; i += 256)
        sw[i] = (_Float16)w1[i];
    __syncthreads();

    const int wave = threadIdx.x >> 5;            // 0..7
    const int lane = threadIdx.x & 31;
    const int tile = blockIdx.x * 8 + wave;       // 16-row tile index
    const int m    = lane & 15;                   // A row / B col within tile
    const int g    = lane >> 4;                   // half-wave id

    if (tile * 16 >= n_nodes) return;             // wave-uniform guard

    const float*    xrow = x  + (size_t)(tile * 16 + m) * 512;
    const _Float16* brow = sw + m * 512;          // B[k][n] = W1[n][k], n==m

    v8f c = {};
    #pragma unroll 4
    for (int kb = 0; kb < 512; kb += 32) {
        // A fragment (16x32 f16): lane holds K = kb+16g+[0..16), contiguous
        const float4 f0 = *(const float4*)(xrow + kb + 16 * g);
        const float4 f1 = *(const float4*)(xrow + kb + 16 * g + 4);
        const float4 f2 = *(const float4*)(xrow + kb + 16 * g + 8);
        const float4 f3 = *(const float4*)(xrow + kb + 16 * g + 12);
        v16h a;
        a[0]  = (_Float16)f0.x; a[1]  = (_Float16)f0.y;
        a[2]  = (_Float16)f0.z; a[3]  = (_Float16)f0.w;
        a[4]  = (_Float16)f1.x; a[5]  = (_Float16)f1.y;
        a[6]  = (_Float16)f1.z; a[7]  = (_Float16)f1.w;
        a[8]  = (_Float16)f2.x; a[9]  = (_Float16)f2.y;
        a[10] = (_Float16)f2.z; a[11] = (_Float16)f2.w;
        a[12] = (_Float16)f3.x; a[13] = (_Float16)f3.y;
        a[14] = (_Float16)f3.z; a[15] = (_Float16)f3.w;
        // B fragment (32x16 f16): lane n holds K = kb+16g+[0..16), contiguous
        const v16h b = *(const v16h*)(brow + kb + 16 * g);
        c = __builtin_amdgcn_wmma_f32_16x16x32_f16(
                false, a, false, b, (short)0, c, false, false);
    }

    // C layout: VGPR v -> row = v + 8*g, col = m
    const float bias = b1[m];
    #pragma unroll
    for (int v = 0; v < 8; ++v) {
        const int r = tile * 16 + v + 8 * g;
        const float val = c[v] + bias;
        h0[(size_t)r * 16 + m] = val > 0.f ? val : 0.f;
    }
}

// ---------------------------------------------------------------------------
// Kernel 2: per-node L2 normalize (16 features)
// ---------------------------------------------------------------------------
__global__ void __launch_bounds__(256) k_normalize(
    const float* __restrict__ h, float* __restrict__ xn, int n)
{
    const int i = blockIdx.x * 256 + threadIdx.x;
    if (i >= n) return;
    const float4* p = (const float4*)(h + (size_t)i * 16);
    float4 a = p[0], b = p[1], c = p[2], d = p[3];
    float s = a.x*a.x + a.y*a.y + a.z*a.z + a.w*a.w
            + b.x*b.x + b.y*b.y + b.z*b.z + b.w*b.w
            + c.x*c.x + c.y*c.y + c.z*c.z + c.w*c.w
            + d.x*d.x + d.y*d.y + d.z*d.z + d.w*d.w;
    const float inv = 1.0f / fmaxf(sqrtf(s), 1e-12f);
    float4* o = (float4*)(xn + (size_t)i * 16);
    a.x*=inv; a.y*=inv; a.z*=inv; a.w*=inv;
    b.x*=inv; b.y*=inv; b.z*=inv; b.w*=inv;
    c.x*=inv; c.y*=inv; c.z*=inv; c.w*=inv;
    d.x*=inv; d.y*=inv; d.z*=inv; d.w*=inv;
    o[0] = a; o[1] = b; o[2] = c; o[3] = d;
}

// ---------------------------------------------------------------------------
// Kernel 3: zero fill
// ---------------------------------------------------------------------------
__global__ void __launch_bounds__(256) k_zero(float* __restrict__ p, size_t count)
{
    const size_t i = (size_t)blockIdx.x * 256 + threadIdx.x;
    if (i < count) p[i] = 0.0f;
}

// ---------------------------------------------------------------------------
// Kernel 4: per-edge exp(beta * cos) + segment-sum denominator.
// Half-wave (16 lanes) per edge; coalesced 64B gathers; shuffle reduce.
// ---------------------------------------------------------------------------
__global__ void __launch_bounds__(256) k_edge_expsum(
    const float* __restrict__ xn,
    const long long* __restrict__ ei,   // [2,E] int64
    long long E, long long ET,          // ET = E + n (self loops appended)
    const float* __restrict__ beta_ptr, // nullptr -> beta = 1
    float* __restrict__ exbuf,          // [ET]
    float* __restrict__ denom)          // [n], pre-zeroed
{
    const int       c16 = threadIdx.x & 15;
    const long long e   = (long long)blockIdx.x * 16 + (threadIdx.x >> 4);
    if (e >= ET) return;

    if (c16 == 0 && e + 4096 < E) {
        __builtin_prefetch(ei + e + 4096, 0, 1);       // global_prefetch_b8
        __builtin_prefetch(ei + E + e + 4096, 0, 1);
    }

    int s, d;
    if (e < E) { s = (int)ei[e]; d = (int)ei[E + e]; }
    else       { s = d = (int)(e - E); }

    float a = xn[(size_t)s * 16 + c16] * xn[(size_t)d * 16 + c16];
    a += __shfl_xor(a, 8);
    a += __shfl_xor(a, 4);
    a += __shfl_xor(a, 2);
    a += __shfl_xor(a, 1);

    const float beta = beta_ptr ? beta_ptr[0] : 1.0f;
    const float ex = __expf(beta * a);
    if (c16 == 0) {
        exbuf[e] = ex;
        atomicAdd(denom + d, ex);
    }
}

// ---------------------------------------------------------------------------
// Kernel 5: weighted scatter-add aggregation: out[d] += (ex/denom[d]) * feat[s]
// ---------------------------------------------------------------------------
__global__ void __launch_bounds__(256) k_edge_aggregate(
    const float* __restrict__ feat,     // un-normalized features [n,16]
    const long long* __restrict__ ei,
    long long E, long long ET,
    const float* __restrict__ exbuf,
    const float* __restrict__ denom,
    float* __restrict__ out)            // [n,16], pre-zeroed
{
    const int       c16 = threadIdx.x & 15;
    const long long e   = (long long)blockIdx.x * 16 + (threadIdx.x >> 4);
    if (e >= ET) return;

    if (c16 == 0 && e + 4096 < E) {
        __builtin_prefetch(ei + e + 4096, 0, 1);
        __builtin_prefetch(ei + E + e + 4096, 0, 1);
    }

    int s, d;
    if (e < E) { s = (int)ei[e]; d = (int)ei[E + e]; }
    else       { s = d = (int)(e - E); }

    const float w = exbuf[e] / denom[d];
    atomicAdd(out + (size_t)d * 16 + c16, w * feat[(size_t)s * 16 + c16]);
}

// ---------------------------------------------------------------------------
// Kernel 6: logits = h @ W2^T + b2, then log_softmax over 8 classes
// ---------------------------------------------------------------------------
__global__ void __launch_bounds__(256) k_lin2_logsoftmax(
    const float* __restrict__ h,    // [n,16]
    const float* __restrict__ w2,   // [8,16]
    const float* __restrict__ b2,   // [8]
    float* __restrict__ out,        // [n,8]
    int n)
{
    __shared__ float sw[8 * 16 + 8];
    if (threadIdx.x < 136)
        sw[threadIdx.x] = (threadIdx.x < 128) ? w2[threadIdx.x]
                                              : b2[threadIdx.x - 128];
    __syncthreads();

    const int i = blockIdx.x * 256 + threadIdx.x;
    if (i >= n) return;

    float hv[16];
    const float4* p = (const float4*)(h + (size_t)i * 16);
    #pragma unroll
    for (int q = 0; q < 4; ++q) {
        const float4 f = p[q];
        hv[4*q+0] = f.x; hv[4*q+1] = f.y; hv[4*q+2] = f.z; hv[4*q+3] = f.w;
    }

    float l[8];
    #pragma unroll
    for (int j = 0; j < 8; ++j) {
        float acc = sw[128 + j];
        #pragma unroll
        for (int k = 0; k < 16; ++k) acc += hv[k] * sw[j * 16 + k];
        l[j] = acc;
    }

    float m = l[0];
    #pragma unroll
    for (int j = 1; j < 8; ++j) m = fmaxf(m, l[j]);
    float ssum = 0.f;
    #pragma unroll
    for (int j = 0; j < 8; ++j) ssum += __expf(l[j] - m);
    const float lse = m + __logf(ssum);

    float4 o0, o1;
    o0.x = l[0]-lse; o0.y = l[1]-lse; o0.z = l[2]-lse; o0.w = l[3]-lse;
    o1.x = l[4]-lse; o1.y = l[5]-lse; o1.z = l[6]-lse; o1.w = l[7]-lse;
    float4* op = (float4*)(out + (size_t)i * 8);
    op[0] = o0; op[1] = o1;
}

// ---------------------------------------------------------------------------
extern "C" void kernel_launch(void* const* d_in, const int* in_sizes, int n_in,
                              void* d_out, int out_size, void* d_ws, size_t ws_size,
                              hipStream_t stream)
{
    const float*     x    = (const float*)d_in[0];          // [n,512]
    const long long* ei   = (const long long*)d_in[1];      // [2,E] int64
    const float*     w1   = (const float*)d_in[2];          // [16,512]
    const float*     b1   = (const float*)d_in[3];          // [16]
    const float*     beta = (const float*)d_in[4];          // [1]
    const float*     w2   = (const float*)d_in[5];          // [8,16]
    const float*     b2   = (const float*)d_in[6];          // [8]
    float*           out  = (float*)d_out;

    const int       n  = in_sizes[0] / 512;
    const long long E  = (long long)in_sizes[1] / 2;
    const long long ET = E + n;

    // workspace layout (floats): hA[n*16] hB[n*16] xn[n*16] denom[n] exbuf[ET]
    float* hA    = (float*)d_ws;
    float* hB    = hA + (size_t)n * 16;
    float* xn    = hB + (size_t)n * 16;
    float* denom = xn + (size_t)n * 16;
    float* exbuf = denom + n;

    const int tiles      = (n + 15) / 16;
    const int lin1_blks  = (tiles + 7) / 8;
    const int node_blks  = (n + 255) / 256;
    const int nf_blks    = (int)(((size_t)n * 16 + 255) / 256);
    const int edge_blks  = (int)((ET + 15) / 16);

    // ---- lin1 + relu (WMMA) ------------------------------------------------
    k_lin1_wmma<<<lin1_blks, 256, 0, stream>>>(x, w1, b1, hA, n);

    // ---- AGNN layer 1 (beta = 1) ------------------------------------------
    k_normalize<<<node_blks, 256, 0, stream>>>(hA, xn, n);
    k_zero<<<node_blks, 256, 0, stream>>>(denom, (size_t)n);
    k_zero<<<nf_blks,   256, 0, stream>>>(hB, (size_t)n * 16);
    k_edge_expsum<<<edge_blks, 256, 0, stream>>>(xn, ei, E, ET, nullptr,
                                                 exbuf, denom);
    k_edge_aggregate<<<edge_blks, 256, 0, stream>>>(hA, ei, E, ET,
                                                    exbuf, denom, hB);

    // ---- AGNN layer 2 (beta = beta2[0]) -----------------------------------
    k_normalize<<<node_blks, 256, 0, stream>>>(hB, xn, n);
    k_zero<<<node_blks, 256, 0, stream>>>(denom, (size_t)n);
    k_zero<<<nf_blks,   256, 0, stream>>>(hA, (size_t)n * 16);
    k_edge_expsum<<<edge_blks, 256, 0, stream>>>(xn, ei, E, ET, beta,
                                                 exbuf, denom);
    k_edge_aggregate<<<edge_blks, 256, 0, stream>>>(hB, ei, E, ET,
                                                    exbuf, denom, hA);

    // ---- lin2 + log_softmax ------------------------------------------------
    k_lin2_logsoftmax<<<node_blks, 256, 0, stream>>>(hA, w2, b2, out, n);
}
